// MultiHeadAttention_77017353552195
// MI455X (gfx1250) — matrix operand order repaired and checked by reading, not compiled, not run on hardware
//
#include <hip/hip_runtime.h>
#include <hip/hip_bf16.h>

#define D_MODEL 1024
#define NUM_HEADS 16
#define DK 64
#define SEQ 2048
#define BATCH 4
#define M_TOT (BATCH * SEQ) // 8192

typedef __attribute__((ext_vector_type(16))) __bf16 v16bf;
typedef __attribute__((ext_vector_type(8)))  __bf16 v8bf;
typedef __attribute__((ext_vector_type(8)))  float  v8f;

union V16U { v16bf v; v8bf h[2]; };

__device__ __forceinline__ unsigned short f2bf(float f) {
    unsigned int u = __float_as_uint(f);
    u += 0x7FFFu + ((u >> 16) & 1u);   // round-to-nearest-even
    return (unsigned short)(u >> 16);
}

__device__ __forceinline__ v8f wmma_bf16(v16bf a, v16bf b, v8f c) {
    // v_wmma_f32_16x16x32_bf16: D = A(16x32) * B(32x16) + C(16x16 f32)
    return __builtin_amdgcn_wmma_f32_16x16x32_bf16(
        false, a, false, b, (short)0, c, false, false);
}

__device__ __forceinline__ v8f zero8() {
    v8f z = {0.f,0.f,0.f,0.f,0.f,0.f,0.f,0.f};
    return z;
}

// A-fragment (16x32 bf16, row-major source): lane half 0 holds K {0..15},
// half 1 holds K {16..31} of row (lane&15).  Two 16B loads.
__device__ __forceinline__ v16bf load_a_frag(const unsigned short* row, int k0, int hl) {
    V16U u;
    u.h[0] = *(const v8bf*)(row + k0 + hl * 16);
    u.h[1] = *(const v8bf*)(row + k0 + hl * 16 + 8);
    return u.v;
}

// ---------------- conversion kernels ----------------

__global__ __launch_bounds__(256) void cvt_x_kernel(
        const float* __restrict__ x, unsigned short* __restrict__ xb, int n) {
    int i = blockIdx.x * blockDim.x + threadIdx.x;
    int stride = gridDim.x * blockDim.x;
    for (; i < n; i += stride) xb[i] = f2bf(x[i]);
}

// Wt[n][k] = bf16(W[k][n])  (so WMMA B-fragments are contiguous per lane)
__global__ __launch_bounds__(256) void cvt_wT_kernel(
        const float* __restrict__ w0, const float* __restrict__ w1,
        const float* __restrict__ w2, const float* __restrict__ w3,
        unsigned short* __restrict__ o0, unsigned short* __restrict__ o1,
        unsigned short* __restrict__ o2, unsigned short* __restrict__ o3) {
    const float* w; unsigned short* o;
    switch (blockIdx.z) {
        case 0:  w = w0; o = o0; break;
        case 1:  w = w1; o = o1; break;
        case 2:  w = w2; o = o2; break;
        default: w = w3; o = o3; break;
    }
    int k = blockIdx.x * 16 + threadIdx.x;
    int n = blockIdx.y * 16 + threadIdx.y;
    o[(size_t)n * D_MODEL + k] = f2bf(w[(size_t)k * D_MODEL + n]);
}

// ---------------- QKV projection GEMM ----------------
// block = 128 thr (4 waves).  All waves share ONE 16-row m-tile (A-fragment
// addresses identical across waves -> WGP$ broadcast); each wave owns a
// 64-col n-strip (4 n-tiles).  Ping-pong double buffering via manual 2x
// unroll (constant buffer indices -> no register rotation moves).
// z = 0:Q  1:K  2:V.  Q,K stored [B,H,T,Dk]; V stored transposed [B,H,Dk,T].
__global__ __launch_bounds__(128) void gemm_qkv_kernel(
        const unsigned short* __restrict__ xb,
        const unsigned short* __restrict__ wqT, const unsigned short* __restrict__ wkT,
        const unsigned short* __restrict__ wvT,
        const float* __restrict__ bq, const float* __restrict__ bk,
        const float* __restrict__ bv,
        unsigned short* __restrict__ qo, unsigned short* __restrict__ ko,
        unsigned short* __restrict__ vto) {
    const int lane = threadIdx.x & 31;
    const int wave = threadIdx.x >> 5;
    const int l15  = lane & 15;
    const int hl   = lane >> 4;
    const int z    = blockIdx.z;

    const unsigned short* wt   = (z == 0) ? wqT : (z == 1) ? wkT : wvT;
    const float*          bias = (z == 0) ? bq  : (z == 1) ? bk  : bv;

    const int m0  = blockIdx.x * 16;                  // shared by all 4 waves
    const int n0b = blockIdx.y * 256 + wave * 64;     // per-wave n-strip

    const unsigned short* xrow = xb + (size_t)(m0 + l15) * D_MODEL;

    auto ldB = [&](int k0, int nt) -> v16bf {
        return *(const v16bf*)(wt + (size_t)(n0b + nt * 16 + l15) * D_MODEL
                                  + k0 + hl * 16);
    };

    v8f acc[4];
#pragma unroll
    for (int i = 0; i < 4; ++i) acc[i] = zero8();

    // ping-pong fragment sets (constant indices -> no rotation moves)
    v16bf aF0, aF1, bF0[4], bF1[4];

    aF0 = load_a_frag(xrow, 0, hl);
#pragma unroll
    for (int nt = 0; nt < 4; ++nt) bF0[nt] = ldB(0, nt);

    for (int k0 = 0; k0 < D_MODEL; k0 += 64) {
        // stage A: prefetch k0+32 into set 1, compute k0 from set 0
        const int k1 = k0 + 32;
        __builtin_prefetch(xrow + k1 + 128);
        aF1 = load_a_frag(xrow, k1, hl);
#pragma unroll
        for (int nt = 0; nt < 4; ++nt) bF1[nt] = ldB(k1, nt);
#pragma unroll
        for (int nt = 0; nt < 4; ++nt)
            acc[nt] = wmma_bf16(aF0, bF0[nt], acc[nt]);

        // stage B: prefetch k0+64 (wraps to 0 at end) into set 0, compute k1
        const int k2 = (k0 + 64) & (D_MODEL - 1);
        aF0 = load_a_frag(xrow, k2, hl);
#pragma unroll
        for (int nt = 0; nt < 4; ++nt) bF0[nt] = ldB(k2, nt);
#pragma unroll
        for (int nt = 0; nt < 4; ++nt)
            acc[nt] = wmma_bf16(aF1, bF1[nt], acc[nt]);
    }

#pragma unroll
    for (int nt = 0; nt < 4; ++nt) {
        const int n = n0b + nt * 16 + l15;
        const float bs = bias[n];
        const int h = n >> 6, d = n & 63;
#pragma unroll
        for (int j = 0; j < 8; ++j) {
            const int m = m0 + hl * 8 + j;          // C-layout row
            const int b = m >> 11, t = m & 2047;
            const unsigned short bf = f2bf(acc[nt][j] + bs);
            if (z == 0)
                qo[(((size_t)(b * NUM_HEADS + h) * SEQ + t) * DK) + d] = bf;
            else if (z == 1)
                ko[(((size_t)(b * NUM_HEADS + h) * SEQ + t) * DK) + d] = bf;
            else
                vto[(((size_t)(b * NUM_HEADS + h) * DK + d) * SEQ) + t] = bf;
        }
    }
}

// ---------------- flash attention ----------------
// grid: x = T/64 (4 waves x 16 query rows), y = B*H.  block = 128.
// The 32-key x 64-d K chunk (4 KB, contiguous) is shared by all 4 waves, so it
// is staged cooperatively into LDS with double-buffered ASYNC loads
// (global_load_async_to_lds_b128 / s_wait_asynccnt) and consumed via ds loads.
// V^T fragments are issued at the top of each iteration so they are in flight
// during the S WMMAs + softmax.
__global__ __launch_bounds__(128) void attn_kernel(
        const unsigned short* __restrict__ q,   // [B,H,T,Dk] bf16
        const unsigned short* __restrict__ k,   // [B,H,T,Dk] bf16
        const unsigned short* __restrict__ vt,  // [B,H,Dk,T] bf16
        unsigned short* __restrict__ ctx) {     // [B*T, C]   bf16
    __shared__ __align__(64) unsigned short kbuf[2][32 * DK]; // 2 x 4KB K chunks
    __shared__ __align__(64) unsigned short pbuf[4][16 * 32]; // wave-private P

    const int tid  = threadIdx.x;
    const int lane = tid & 31;
    const int wave = tid >> 5;
    const int l15  = lane & 15;
    const int hl   = lane >> 4;
    const int bh   = blockIdx.y;
    const int t0   = blockIdx.x * 64 + wave * 16;

    const unsigned short* qb = q  + (size_t)bh * SEQ * DK;
    const unsigned short* kb = k  + (size_t)bh * SEQ * DK;
    const unsigned short* vb = vt + (size_t)bh * DK * SEQ;

    const unsigned short* qrow = qb + (size_t)(t0 + l15) * DK;
    const v16bf aq0 = load_a_frag(qrow, 0,  hl);
    const v16bf aq1 = load_a_frag(qrow, 32, hl);

    v8f o[4];
    float mrow[8], lrow[8];
#pragma unroll
    for (int i = 0; i < 4; ++i) o[i] = zero8();
#pragma unroll
    for (int j = 0; j < 8; ++j) { mrow[j] = -3.0e38f; lrow[j] = 0.f; }

    unsigned short* myp = pbuf[wave];
    const float scale = 0.125f; // 1/sqrt(64)

    // LDS byte offsets of the two K staging buffers (low 32 bits of flat addr)
    const unsigned int kofs0 = (unsigned int)(unsigned long long)(uintptr_t)&kbuf[0][0];
    const unsigned int kofs1 = (unsigned int)(unsigned long long)(uintptr_t)&kbuf[1][0];

    // cooperative async stage of K rows [kk, kk+32) into buffer `buf`:
    // chunk is 4096 B contiguous; 128 threads x 2 x 16B.
    auto stage_k = [&](int kk, int buf) {
        const unsigned int base = (buf ? kofs1 : kofs0) + tid * 16;
        unsigned long long ga =
            (unsigned long long)(uintptr_t)(kb + (size_t)kk * DK) + tid * 16;
        asm volatile("global_load_async_to_lds_b128 %0, %1, off"
                     :: "v"(base), "v"(ga) : "memory");
        asm volatile("global_load_async_to_lds_b128 %0, %1, off"
                     :: "v"(base + 2048u), "v"(ga + 2048ull) : "memory");
    };

    stage_k(0, 0);

    for (int kk = 0; kk < SEQ; kk += 32) {
        const int buf = (kk >> 5) & 1;
        if (kk + 32 < SEQ) {
            stage_k(kk + 32, buf ^ 1);
            // 2 newest ops outstanding -> current chunk's 2 ops complete
            asm volatile("s_wait_asynccnt 0x2" ::: "memory");
        } else {
            asm volatile("s_wait_asynccnt 0x0" ::: "memory");
        }
        __syncthreads();  // all waves' staged portions visible

        // Issue V^T B-fragments early; consumed after softmax.
        v16bf bvf[4];
#pragma unroll
        for (int dt = 0; dt < 4; ++dt)
            bvf[dt] = *(const v16bf*)(vb + (size_t)(dt * 16 + l15) * SEQ
                                         + kk + hl * 16);

        const unsigned short* kc = kbuf[buf];

        // S = Q * K^T : two 16x16 tiles (keys kk..+15, kk+16..+31), from LDS
        v8f s0 = zero8(), s1 = zero8();
        {
            const unsigned short* kr0 = kc + (size_t)l15 * DK;
            const unsigned short* kr1 = kc + (size_t)(16 + l15) * DK;
            v16bf b00 = *(const v16bf*)(kr0 + hl * 16);
            v16bf b01 = *(const v16bf*)(kr0 + 32 + hl * 16);
            v16bf b10 = *(const v16bf*)(kr1 + hl * 16);
            v16bf b11 = *(const v16bf*)(kr1 + 32 + hl * 16);
            s0 = wmma_bf16(aq0, b00, s0);
            s0 = wmma_bf16(aq1, b01, s0);
            s1 = wmma_bf16(aq0, b10, s1);
            s1 = wmma_bf16(aq1, b11, s1);
        }

        // online softmax; row r = hl*8+j lives across the 16 lanes of this half
#pragma unroll
        for (int j = 0; j < 8; ++j) {
            float a  = s0[j] * scale;
            float b2 = s1[j] * scale;
            float cm = fmaxf(a, b2);
            cm = fmaxf(cm, __shfl_xor(cm, 1, 32));
            cm = fmaxf(cm, __shfl_xor(cm, 2, 32));
            cm = fmaxf(cm, __shfl_xor(cm, 4, 32));
            cm = fmaxf(cm, __shfl_xor(cm, 8, 32));
            float mn = fmaxf(mrow[j], cm);
            float p0 = __expf(a  - mn);
            float p1 = __expf(b2 - mn);
            float rs = p0 + p1;
            rs += __shfl_xor(rs, 1, 32);
            rs += __shfl_xor(rs, 2, 32);
            rs += __shfl_xor(rs, 4, 32);
            rs += __shfl_xor(rs, 8, 32);
            float corr = __expf(mrow[j] - mn);
            lrow[j] = lrow[j] * corr + rs;
            mrow[j] = mn;
            o[0][j] *= corr; o[1][j] *= corr; o[2][j] *= corr; o[3][j] *= corr;
            const int r = hl * 8 + j;
            myp[r * 32 + l15]      = f2bf(p0);
            myp[r * 32 + 16 + l15] = f2bf(p1);
        }

        // C-layout -> A-layout via LDS (wave-private region)
        V16U ap;
        ap.h[0] = *(const v8bf*)(myp + l15 * 32 + hl * 16);
        ap.h[1] = *(const v8bf*)(myp + l15 * 32 + hl * 16 + 8);

        // O += P * V
#pragma unroll
        for (int dt = 0; dt < 4; ++dt)
            o[dt] = wmma_bf16(ap.v, bvf[dt], o[dt]);

        __syncthreads();  // readers done before next chunk overwrites buffer
    }

    const int b = bh >> 4, h = bh & 15;
#pragma unroll
    for (int j = 0; j < 8; ++j) {
        const float inv = 1.0f / lrow[j];
        const size_t gt = (size_t)b * SEQ + t0 + hl * 8 + j;
#pragma unroll
        for (int dt = 0; dt < 4; ++dt) {
            ctx[gt * D_MODEL + h * 64 + dt * 16 + l15] = f2bf(o[dt][j] * inv);
        }
    }
}

// ---------------- output projection GEMM ----------------
__global__ __launch_bounds__(128) void gemm_out_kernel(
        const unsigned short* __restrict__ cx,  // [8192,1024] bf16
        const unsigned short* __restrict__ woT, // [out,in] bf16
        const float* __restrict__ bo,
        float* __restrict__ out) {
    const int lane = threadIdx.x & 31;
    const int wave = threadIdx.x >> 5;
    const int l15  = lane & 15;
    const int hl   = lane >> 4;

    const int m0  = blockIdx.x * 16;               // shared by all 4 waves
    const int n0b = blockIdx.y * 256 + wave * 64;  // per-wave n-strip

    const unsigned short* xrow = cx + (size_t)(m0 + l15) * D_MODEL;

    auto ldB = [&](int k0, int nt) -> v16bf {
        return *(const v16bf*)(woT + (size_t)(n0b + nt * 16 + l15) * D_MODEL
                                   + k0 + hl * 16);
    };

    v8f acc[4];
#pragma unroll
    for (int i = 0; i < 4; ++i) acc[i] = zero8();

    v16bf aF0, aF1, bF0[4], bF1[4];

    aF0 = load_a_frag(xrow, 0, hl);
#pragma unroll
    for (int nt = 0; nt < 4; ++nt) bF0[nt] = ldB(0, nt);

    for (int k0 = 0; k0 < D_MODEL; k0 += 64) {
        const int k1 = k0 + 32;
        __builtin_prefetch(xrow + k1 + 128);
        aF1 = load_a_frag(xrow, k1, hl);
#pragma unroll
        for (int nt = 0; nt < 4; ++nt) bF1[nt] = ldB(k1, nt);
#pragma unroll
        for (int nt = 0; nt < 4; ++nt)
            acc[nt] = wmma_bf16(aF0, bF0[nt], acc[nt]);

        const int k2 = (k0 + 64) & (D_MODEL - 1);
        aF0 = load_a_frag(xrow, k2, hl);
#pragma unroll
        for (int nt = 0; nt < 4; ++nt) bF0[nt] = ldB(k2, nt);
#pragma unroll
        for (int nt = 0; nt < 4; ++nt)
            acc[nt] = wmma_bf16(aF1, bF1[nt], acc[nt]);
    }

#pragma unroll
    for (int nt = 0; nt < 4; ++nt) {
        const int n = n0b + nt * 16 + l15;
        const float bs = bo[n];
#pragma unroll
        for (int j = 0; j < 8; ++j) {
            const int m = m0 + hl * 8 + j;
            out[(size_t)m * D_MODEL + n] = acc[nt][j] + bs;
        }
    }
}

// ---------------- host launcher ----------------
extern "C" void kernel_launch(void* const* d_in, const int* in_sizes, int n_in,
                              void* d_out, int out_size, void* d_ws, size_t ws_size,
                              hipStream_t stream) {
    const float* x  = (const float*)d_in[0];
    const float* Wq = (const float*)d_in[1];
    const float* bq = (const float*)d_in[2];
    const float* Wk = (const float*)d_in[3];
    const float* bk = (const float*)d_in[4];
    const float* Wv = (const float*)d_in[5];
    const float* bv = (const float*)d_in[6];
    const float* Wo = (const float*)d_in[7];
    const float* bo = (const float*)d_in[8];
    float* out = (float*)d_out;

    char* ws = (char*)d_ws;
    size_t off = 0;
    auto alloc = [&](size_t bytes) -> char* {
        char* p = ws + off;
        off += (bytes + 255) & ~(size_t)255;
        return p;
    };
    const size_t act_b = (size_t)M_TOT * D_MODEL * sizeof(unsigned short);   // 16 MB
    const size_t w_b   = (size_t)D_MODEL * D_MODEL * sizeof(unsigned short); //  2 MB

    unsigned short* xb  = (unsigned short*)alloc(act_b);
    unsigned short* wqT = (unsigned short*)alloc(w_b);
    unsigned short* wkT = (unsigned short*)alloc(w_b);
    unsigned short* wvT = (unsigned short*)alloc(w_b);
    unsigned short* woT = (unsigned short*)alloc(w_b);
    unsigned short* qbf = (unsigned short*)alloc(act_b);
    unsigned short* kbf = (unsigned short*)alloc(act_b);
    unsigned short* vtb = (unsigned short*)alloc(act_b);
    unsigned short* ctx = (unsigned short*)alloc(act_b);
    if (off > ws_size) return; // ~88 MB needed

    cvt_x_kernel<<<4096, 256, 0, stream>>>(x, xb, M_TOT * D_MODEL);
    cvt_wT_kernel<<<dim3(64, 64, 4), dim3(16, 16), 0, stream>>>(
        Wq, Wk, Wv, Wo, wqT, wkT, wvT, woT);
    gemm_qkv_kernel<<<dim3(512, 4, 3), 128, 0, stream>>>(
        xb, wqT, wkT, wvT, bq, bk, bv, qbf, kbf, vtb);
    attn_kernel<<<dim3(SEQ / 64, BATCH * NUM_HEADS), 128, 0, stream>>>(
        qbf, kbf, vtb, ctx);
    gemm_out_kernel<<<dim3(512, 4), 128, 0, stream>>>(ctx, woT, bo, out);
}
